// TransformerEncoder_59889023976128
// MI455X (gfx1250) — compile-verified, matrix operand order checked by reading, not run on hardware
//
#include <hip/hip_runtime.h>

#define DIMC   512
#define HEADS_C 8
#define MLP    2048
#define DEPTH_C 4
#define BATCH  2
#define SEQ    2048
#define M_TOT  (BATCH*SEQ)   // 4096 rows in the token dimension

typedef __attribute__((ext_vector_type(16))) _Float16 v16h;
typedef __attribute__((ext_vector_type(8)))  _Float16 v8h;
typedef __attribute__((ext_vector_type(8)))  float    v8f;

// ---------------------------------------------------------------------------
// WMMA helpers
// ---------------------------------------------------------------------------
__device__ __forceinline__ v8f wmma_f16(v16h a, v16h b, v8f c) {
  // D = A(16x32 f16) x B(32x16 f16) + C(16x16 f32)
  return __builtin_amdgcn_wmma_f32_16x16x32_f16(false, a, false, b,
                                                (short)0, c, false, false);
}

// Load a 16x32 f16 operand fragment (A layout; also serves B when the source
// is stored "column-major as rows", i.e. Wt[N,K] or Vt[dim,key]).
// base points at (row0, k0); stride in elements. Per-lane:
//   lanes 0-15 : row = lane,     K = k0+0..7   and k0+16..23
//   lanes 16-31: row = lane-16,  K = k0+8..15  and k0+24..31
__device__ __forceinline__ v16h load_frag(const _Float16* base, int stride, int lane) {
  const _Float16* p = base + (size_t)(lane & 15) * stride + ((lane >> 4) << 3);
  v8h lo = *(const v8h*)(p);
  v8h hi = *(const v8h*)(p + 16);
  v16h r;
#pragma unroll
  for (int i = 0; i < 8; ++i) { r[i] = lo[i]; r[i + 8] = hi[i]; }
  return r;
}

// ---------------------------------------------------------------------------
// Elementwise converts
// ---------------------------------------------------------------------------
__global__ void k_f32_to_f16(const float* __restrict__ in, _Float16* __restrict__ out, int n) {
  int i = blockIdx.x * blockDim.x + threadIdx.x;
  if (i < n) out[i] = (_Float16)in[i];
}

// w: [K,N] f32 row-major  ->  wt: [N,K] f16 row-major (transposed)
__global__ void k_transpose_f16(const float* __restrict__ w, _Float16* __restrict__ wt,
                                int K, int N) {
  int i = blockIdx.x * blockDim.x + threadIdx.x;
  if (i < K * N) {
    int n = i / K, k = i - n * K;
    wt[i] = (_Float16)w[(size_t)k * N + n];
  }
}

// ---------------------------------------------------------------------------
// LayerNorm (row = 512 floats), one wave per row, f16 output for WMMA A operand
// ---------------------------------------------------------------------------
__global__ __launch_bounds__(128) void k_layernorm_f16(const float* __restrict__ x,
                                                       const float* __restrict__ g,
                                                       const float* __restrict__ bb,
                                                       _Float16* __restrict__ out) {
  int wave = threadIdx.x >> 5, lane = threadIdx.x & 31;
  int row  = blockIdx.x * 4 + wave;
  const float* xr = x + (size_t)row * DIMC;
  float v[16], s = 0.f, s2 = 0.f;
#pragma unroll
  for (int i = 0; i < 16; ++i) {
    float t = xr[lane + i * 32];
    v[i] = t; s += t; s2 += t * t;
  }
#pragma unroll
  for (int off = 1; off < 32; off <<= 1) {
    s  += __shfl_xor(s,  off, 32);
    s2 += __shfl_xor(s2, off, 32);
  }
  float mu  = s * (1.f / DIMC);
  float var = s2 * (1.f / DIMC) - mu * mu;
  float rs  = rsqrtf(var + 1e-5f);
  _Float16* o = out + (size_t)row * DIMC;
#pragma unroll
  for (int i = 0; i < 16; ++i) {
    int c = lane + i * 32;
    o[c] = (_Float16)((v[i] - mu) * rs * g[c] + bb[c]);
  }
}

// ---------------------------------------------------------------------------
// WMMA GEMM: C[M,N] = A[M,K](f16) * Wt[N,K]^T(f16) (+bias)(+gelu)(+resid)
// wave tile 32x64, block tile 32x256, 4 waves
// flags: 1 = exact GELU, 2 = write f16 output (else f32)
// ---------------------------------------------------------------------------
__global__ __launch_bounds__(128) void k_gemm(const _Float16* __restrict__ A,
                                              const _Float16* __restrict__ Bt,
                                              const float* __restrict__ bias,
                                              const float* __restrict__ resid,
                                              void* __restrict__ Cout,
                                              int M, int N, int K, int flags) {
  int wave = threadIdx.x >> 5, lane = threadIdx.x & 31;
  int m0 = blockIdx.y * 32;
  int n0 = blockIdx.x * 256 + wave * 64;
  if (m0 >= M || n0 >= N) return;

  v8f acc[2][4] = {};
  for (int kb = 0; kb < K; kb += 32) {
    v16h a0 = load_frag(A + (size_t)m0 * K + kb,        K, lane);
    v16h a1 = load_frag(A + (size_t)(m0 + 16) * K + kb, K, lane);
#pragma unroll
    for (int t = 0; t < 4; ++t) {
      v16h bf = load_frag(Bt + (size_t)(n0 + t * 16) * K + kb, K, lane);
      acc[0][t] = wmma_f16(a0, bf, acc[0][t]);
      acc[1][t] = wmma_f16(a1, bf, acc[1][t]);
    }
  }

  int hp = lane >> 4, nn = lane & 15;
#pragma unroll
  for (int i = 0; i < 2; ++i)
#pragma unroll
    for (int t = 0; t < 4; ++t) {
      int n = n0 + t * 16 + nn;
#pragma unroll
      for (int r = 0; r < 8; ++r) {
        int m = m0 + i * 16 + r + hp * 8;
        float val = acc[i][t][r];
        if (bias)      val += bias[n];
        if (flags & 1) val  = 0.5f * val * (1.f + erff(val * 0.70710678118f));
        if (resid)     val += resid[(size_t)m * N + n];
        if (flags & 2) ((_Float16*)Cout)[(size_t)m * N + n] = (_Float16)val;
        else           ((float*)Cout)[(size_t)m * N + n]    = val;
      }
    }
}

// ---------------------------------------------------------------------------
// Flash attention. qkv: [B, SEQ, 1536] f32 (q|k|v). out16: [B*SEQ, 512] f16.
// grid = (SEQ/64, B*HEADS). Block = 4 waves; each wave owns 16 query rows.
// ---------------------------------------------------------------------------
__global__ __launch_bounds__(128) void k_attention(const float* __restrict__ qkv,
                                                   _Float16* __restrict__ out16) {
  __shared__ __align__(16) _Float16 sK[64 * 64];      // [key][dim]
  __shared__ __align__(16) _Float16 sVt[64 * 64];     // [dim][key] (transposed)
  __shared__ __align__(16) _Float16 sQ[4][16 * 64];   // per-wave Q tile
  __shared__ __align__(16) _Float16 sP[4][16 * 64];   // per-wave P scratch

  int wave = threadIdx.x >> 5, lane = threadIdx.x & 31;
  int qb = blockIdx.x;
  int bh = blockIdx.y;
  int b  = bh >> 3, h = bh & 7;
  const float* base = qkv + (size_t)b * SEQ * 1536;
  int qrow0 = qb * 64 + wave * 16;

  // stage Q (f32 -> f16), per wave
  for (int idx = lane; idx < 16 * 64; idx += 32) {
    int r = idx >> 6, d = idx & 63;
    sQ[wave][idx] = (_Float16)base[(size_t)(qrow0 + r) * 1536 + h * 64 + d];
  }
  v16h aq0 = load_frag(&sQ[wave][0],  64, lane);
  v16h aq1 = load_frag(&sQ[wave][32], 64, lane);

  v8f o[4] = {};
  float mst[8], lst[8];
#pragma unroll
  for (int r = 0; r < 8; ++r) { mst[r] = -1e30f; lst[r] = 0.f; }
  const float scale = 0.125f;  // 64^-0.5
  int hp = lane >> 4, nn = lane & 15;

  for (int j = 0; j < SEQ / 64; ++j) {
    __syncthreads();
    // stage K (row-major) and V (transposed) cooperatively
    for (int idx = (int)threadIdx.x; idx < 64 * 64; idx += 128) {
      int r = idx >> 6, d = idx & 63;
      const float* kv = base + (size_t)(j * 64 + r) * 1536 + h * 64;
      sK[idx]         = (_Float16)kv[512 + d];
      sVt[d * 64 + r] = (_Float16)kv[1024 + d];
    }
    __syncthreads();

    // S = Q * K^T  (4 tiles of 16x16, K-dim 64 = two chained WMMAs)
    v8f s[4];
#pragma unroll
    for (int t = 0; t < 4; ++t) {
      v8f z = {};
      z = wmma_f16(aq0, load_frag(&sK[t * 16 * 64],      64, lane), z);
      z = wmma_f16(aq1, load_frag(&sK[t * 16 * 64 + 32], 64, lane), z);
      s[t] = z;
    }

    // online softmax per row (rows live at C-element r, halves split by hp)
#pragma unroll
    for (int r = 0; r < 8; ++r) {
      float mx = -1e30f;
#pragma unroll
      for (int t = 0; t < 4; ++t) { float v = s[t][r] * scale; s[t][r] = v; mx = fmaxf(mx, v); }
#pragma unroll
      for (int off = 1; off < 16; off <<= 1) mx = fmaxf(mx, __shfl_xor(mx, off, 32));
      float mnew = fmaxf(mst[r], mx);
      float corr = __expf(mst[r] - mnew);
      float rsum = 0.f;
#pragma unroll
      for (int t = 0; t < 4; ++t) { float p = __expf(s[t][r] - mnew); s[t][r] = p; rsum += p; }
#pragma unroll
      for (int off = 1; off < 16; off <<= 1) rsum += __shfl_xor(rsum, off, 32);
      lst[r] = lst[r] * corr + rsum;
      mst[r] = mnew;
#pragma unroll
      for (int t = 0; t < 4; ++t) o[t][r] *= corr;
    }

    // C-layout P -> LDS -> A-layout fragments
#pragma unroll
    for (int t = 0; t < 4; ++t)
#pragma unroll
      for (int r = 0; r < 8; ++r)
        sP[wave][(r + hp * 8) * 64 + t * 16 + nn] = (_Float16)s[t][r];
    v16h ap0 = load_frag(&sP[wave][0],  64, lane);
    v16h ap1 = load_frag(&sP[wave][32], 64, lane);

    // O += P * V   (B operand from transposed V: contiguous per lane)
#pragma unroll
    for (int t = 0; t < 4; ++t) {
      o[t] = wmma_f16(ap0, load_frag(&sVt[t * 16 * 64],      64, lane), o[t]);
      o[t] = wmma_f16(ap1, load_frag(&sVt[t * 16 * 64 + 32], 64, lane), o[t]);
    }
  }

  // normalize and store heads-concatenated f16 output
#pragma unroll
  for (int r = 0; r < 8; ++r) {
    float inv = 1.f / lst[r];
    int m = b * SEQ + qrow0 + r + hp * 8;
#pragma unroll
    for (int t = 0; t < 4; ++t)
      out16[(size_t)m * DIMC + h * 64 + t * 16 + nn] = (_Float16)(o[t][r] * inv);
  }
}

// ---------------------------------------------------------------------------
// Launch
// ---------------------------------------------------------------------------
extern "C" void kernel_launch(void* const* d_in, const int* in_sizes, int n_in,
                              void* d_out, int out_size, void* d_ws, size_t ws_size,
                              hipStream_t stream) {
  (void)in_sizes; (void)n_in; (void)out_size; (void)ws_size;

  const float* x_in   = (const float*)d_in[0];
  const float* proj_w = (const float*)d_in[1];
  const float* proj_b = (const float*)d_in[2];

  // workspace layout (~35.5 MB)
  char* ws = (char*)d_ws;
  size_t off = 0;
  auto alloc = [&](size_t bytes) { void* p = ws + off; off += (bytes + 255) & ~(size_t)255; return p; };
  _Float16* a16    = (_Float16*)alloc((size_t)M_TOT * DIMC * 2);       // ln / x0 f16
  _Float16* attn16 = (_Float16*)alloc((size_t)M_TOT * DIMC * 2);       // attention out f16
  float*    qkv    = (float*)   alloc((size_t)M_TOT * 3 * DIMC * 4);   // qkv f32
  _Float16* wt     = (_Float16*)alloc((size_t)MLP * DIMC * 2);         // transposed weight f16
  _Float16* h16    = (_Float16*)qkv;  // ff1 activation reuses qkv buffer (dead after attention)

  float* x = (float*)d_out;  // residual stream lives in d_out; fully overwritten by first GEMM

  dim3 blk(128);

  // ---- input projection: x = x_in @ proj_w + proj_b ----
  { int n = M_TOT * DIMC; k_f32_to_f16<<<(n + 255) / 256, 256, 0, stream>>>(x_in, a16, n); }
  { int n = DIMC * DIMC;  k_transpose_f16<<<(n + 255) / 256, 256, 0, stream>>>(proj_w, wt, DIMC, DIMC); }
  k_gemm<<<dim3(DIMC / 256, M_TOT / 32), blk, 0, stream>>>(a16, wt, proj_b, nullptr, x,
                                                           M_TOT, DIMC, DIMC, 0);

  for (int l = 0; l < DEPTH_C; ++l) {
    void* const* p = d_in + 3 + l * 11;
    const float* ln1_g = (const float*)p[0];
    const float* ln1_b = (const float*)p[1];
    const float* qkv_w = (const float*)p[2];
    const float* out_w = (const float*)p[3];
    const float* out_b = (const float*)p[4];
    const float* ln2_g = (const float*)p[5];
    const float* ln2_b = (const float*)p[6];
    const float* ff1_w = (const float*)p[7];
    const float* ff1_b = (const float*)p[8];
    const float* ff2_w = (const float*)p[9];
    const float* ff2_b = (const float*)p[10];

    // attention block
    k_layernorm_f16<<<M_TOT / 4, blk, 0, stream>>>(x, ln1_g, ln1_b, a16);
    { int n = DIMC * 3 * DIMC; k_transpose_f16<<<(n + 255) / 256, 256, 0, stream>>>(qkv_w, wt, DIMC, 3 * DIMC); }
    k_gemm<<<dim3(3 * DIMC / 256, M_TOT / 32), blk, 0, stream>>>(a16, wt, nullptr, nullptr, qkv,
                                                                 M_TOT, 3 * DIMC, DIMC, 0);
    k_attention<<<dim3(SEQ / 64, BATCH * HEADS_C), blk, 0, stream>>>(qkv, attn16);
    { int n = DIMC * DIMC; k_transpose_f16<<<(n + 255) / 256, 256, 0, stream>>>(out_w, wt, DIMC, DIMC); }
    k_gemm<<<dim3(DIMC / 256, M_TOT / 32), blk, 0, stream>>>(attn16, wt, out_b, x, x,
                                                             M_TOT, DIMC, DIMC, 0);

    // MLP block
    k_layernorm_f16<<<M_TOT / 4, blk, 0, stream>>>(x, ln2_g, ln2_b, a16);
    { int n = DIMC * MLP; k_transpose_f16<<<(n + 255) / 256, 256, 0, stream>>>(ff1_w, wt, DIMC, MLP); }
    k_gemm<<<dim3(MLP / 256, M_TOT / 32), blk, 0, stream>>>(a16, wt, ff1_b, nullptr, h16,
                                                            M_TOT, MLP, DIMC, /*gelu+f16out*/ 1 | 2);
    { int n = MLP * DIMC; k_transpose_f16<<<(n + 255) / 256, 256, 0, stream>>>(ff2_w, wt, MLP, DIMC); }
    k_gemm<<<dim3(DIMC / 256, M_TOT / 32), blk, 0, stream>>>(h16, wt, ff2_b, x, x,
                                                             M_TOT, DIMC, MLP, 0);
  }
}
